// SelfAttention_3470333575275
// MI455X (gfx1250) — compile-verified, matrix operand order checked by reading, not run on hardware
//
#include <hip/hip_runtime.h>
#include <hip/hip_bf16.h>

// ---------------------------------------------------------------------------
// Self-attention forward for MI455X (gfx1250, wave32, WMMA).
//   1) f32 -> bf16 casts of x and weights (workspace)
//   2) QKV projection (2x2 register-blocked bf16 WMMA) + bias + RoPE;
//      Q,K stored [B,H,T,D] bf16; V stored transposed [B,H,D,T] bf16
//   3) flash attention: 32 queries/wave, S^T and O^T computed via WMMA so
//      softmax stats & P stay per-lane (no transposes); K/V shared by 2 q-tiles
//   4) output projection (2x2 WMMA blocking) + bias -> f32 d_out
// ---------------------------------------------------------------------------

typedef __attribute__((ext_vector_type(16))) __bf16 v16bf;
typedef __attribute__((ext_vector_type(8)))  __bf16 v8bf;
typedef __attribute__((ext_vector_type(8)))  float  v8f;

#define WMMA_BF16(A, B, C) \
  __builtin_amdgcn_wmma_f32_16x16x32_bf16(false, (A), false, (B), (short)0, (C), false, false)

static __device__ __forceinline__ v8f vzero8() {
  v8f z = {0.f, 0.f, 0.f, 0.f, 0.f, 0.f, 0.f, 0.f};
  return z;
}

// One 16x32 bf16 WMMA operand for this lane from two 16-byte chunks:
// elements 0..7 = K(+0..7), elements 8..15 = K(+16..23) of the lane's chunking.
static __device__ __forceinline__ v16bf mk16(const __bf16* p0, const __bf16* p1) {
  v8bf a = *reinterpret_cast<const v8bf*>(p0);
  v8bf b = *reinterpret_cast<const v8bf*>(p1);
  v16bf r;
#pragma unroll
  for (int e = 0; e < 8; ++e) { r[e] = a[e]; r[8 + e] = b[e]; }
  return r;
}

// ---------------------------------------------------------------------------
// 1) f32 -> bf16, 8 elements per thread
// ---------------------------------------------------------------------------
__global__ void f32_to_bf16_vec8(const float* __restrict__ in,
                                 __bf16* __restrict__ out, int nvec) {
  int i = blockIdx.x * blockDim.x + threadIdx.x;
  if (i >= nvec) return;
  const float4* p = reinterpret_cast<const float4*>(in) + (size_t)i * 2;
  float4 a = p[0], b = p[1];
  v8bf o;
  o[0] = (__bf16)a.x; o[1] = (__bf16)a.y; o[2] = (__bf16)a.z; o[3] = (__bf16)a.w;
  o[4] = (__bf16)b.x; o[5] = (__bf16)b.y; o[6] = (__bf16)b.z; o[7] = (__bf16)b.w;
  reinterpret_cast<v8bf*>(out)[i] = o;
}

// ---------------------------------------------------------------------------
// 2) QKV projection + bias + RoPE.  2x2 tile blocking: each wave computes a
//    32x32 output block (4 WMMAs / 32-k step, operands reused 2x each).
//    grid = (32 n-blocks, 16 m-groups, 3 projs), block = 256 (8 waves).
// ---------------------------------------------------------------------------
__global__ __launch_bounds__(256) void qkv_rope_kernel(
    const __bf16* __restrict__ xb,
    const __bf16* __restrict__ wqb, const __bf16* __restrict__ wkb,
    const __bf16* __restrict__ wvb,
    const float* __restrict__ bq, const float* __restrict__ bk,
    const float* __restrict__ bv,
    __bf16* __restrict__ qr, __bf16* __restrict__ kr, __bf16* __restrict__ vt) {
  const int lane = threadIdx.x & 31;
  const int wave = threadIdx.x >> 5;
  const int proj = blockIdx.z;
  const int mt = blockIdx.y * 8 + wave;          // 0..127 (32-row blocks)
  const int nt = blockIdx.x;                     // 0..31  (32-col blocks)
  const int lq = lane & 15, hi = lane >> 4, koff = hi * 8;
  const int row0 = mt * 32, col0 = nt * 32;

  const __bf16* w   = (proj == 0) ? wqb : (proj == 1) ? wkb : wvb;
  const float*  bia = (proj == 0) ? bq  : (proj == 1) ? bk  : bv;

  const __bf16* arow0 = xb + (size_t)(row0 + lq) * 1024;
  const __bf16* arow1 = arow0 + (size_t)16 * 1024;
  const __bf16* brow0 = w + (size_t)(col0 + lq) * 1024;   // out[m,n]=sum_k x[m,k]w[n,k]
  const __bf16* brow1 = brow0 + (size_t)16 * 1024;

  v8f acc[2][2];
  acc[0][0] = vzero8(); acc[0][1] = vzero8();
  acc[1][0] = vzero8(); acc[1][1] = vzero8();

#pragma unroll 2
  for (int k0 = 0; k0 < 1024; k0 += 32) {
    __builtin_prefetch(arow0 + k0 + 256, 0, 1);
    __builtin_prefetch(brow0 + k0 + 256, 0, 1);
    v16bf A0 = mk16(arow0 + k0 + koff, arow0 + k0 + 16 + koff);
    v16bf A1 = mk16(arow1 + k0 + koff, arow1 + k0 + 16 + koff);
    v16bf B0 = mk16(brow0 + k0 + koff, brow0 + k0 + 16 + koff);
    v16bf B1 = mk16(brow1 + k0 + koff, brow1 + k0 + 16 + koff);
    acc[0][0] = WMMA_BF16(A0, B0, acc[0][0]);
    acc[0][1] = WMMA_BF16(A0, B1, acc[0][1]);
    acc[1][0] = WMMA_BF16(A1, B0, acc[1][0]);
    acc[1][1] = WMMA_BF16(A1, B1, acc[1][1]);
  }

  const int b = row0 >> 11;                       // batch; 2048 % 32 == 0
#pragma unroll
  for (int ni = 0; ni < 2; ++ni) {
    const int   c    = col0 + ni * 16 + lq;       // output channel of this lane
    const float bval = bia[c];
    const int   h    = c >> 6;
    const int   dl   = c & 63;
#pragma unroll
    for (int mi = 0; mi < 2; ++mi) {
      const int rbase = (row0 & 2047) + mi * 16;  // t of tile row 0
      v8f a = acc[mi][ni];
#pragma unroll
      for (int v = 0; v < 8; ++v) a[v] += bval;

      if (proj == 2) {
        // V transposed [B,H,D,T]; VGPRs are consecutive t -> one b128 store
        const int t0 = rbase + hi * 8;
        v8bf pv;
#pragma unroll
        for (int v = 0; v < 8; ++v) pv[v] = (__bf16)a[v];
        *reinterpret_cast<v8bf*>(vt + ((size_t)(b * 16 + h) * 64 + dl) * 2048 + t0) = pv;
      } else {
        // RoPE: interleaved rotate_half; angle = t * 10000^(-(dl%32)/32)
        __bf16* dst = ((proj == 0) ? qr : kr) + (size_t)(b * 16 + h) * 2048 * 64;
        const float inv_freq = __powf(10000.f, -(float)(dl & 31) * (1.f / 32.f));
        const float sgn = (dl & 1) ? 1.f : -1.f;
#pragma unroll
        for (int v = 0; v < 8; ++v) {
          float self    = a[v];
          float partner = __shfl_xor(self, 1, 32);  // adjacent channel = lane^1
          int   t   = rbase + v + hi * 8;
          float ang = (float)t * inv_freq;
          float ro  = self * __cosf(ang) + sgn * partner * __sinf(ang);
          dst[(size_t)t * 64 + dl] = (__bf16)ro;
        }
      }
    }
  }
}

// ---------------------------------------------------------------------------
// online-softmax block update for one q-tile: consumes the two unscaled S^T
// tiles (s0 = keys kb..kb+15, s1 = keys kb+16..31), updates (m_i, l_i),
// returns exp-probabilities packed directly in the 16x32 B-operand layout.
// ---------------------------------------------------------------------------
static __device__ __forceinline__ v16bf softmax_update(v8f s0, v8f s1,
                                                       float& m_i, float& l_i,
                                                       float& alpha) {
  float mb = -3.0e38f;
#pragma unroll
  for (int v = 0; v < 8; ++v) {
    s0[v] *= 0.125f;  s1[v] *= 0.125f;            // 1/sqrt(64)
    mb = fmaxf(mb, fmaxf(s0[v], s1[v]));
  }
  mb = fmaxf(mb, __shfl_xor(mb, 16, 32));         // other 16 keys live in lane^16
  const float mnew = fmaxf(m_i, mb);
  alpha = __expf(m_i - mnew);
  v16bf P;
  float rs = 0.f;
#pragma unroll
  for (int v = 0; v < 8; ++v) {
    float p0 = __expf(s0[v] - mnew);
    float p1 = __expf(s1[v] - mnew);
    rs += p0 + p1;
    P[v] = (__bf16)p0;  P[8 + v] = (__bf16)p1;
  }
  rs += __shfl_xor(rs, 16, 32);
  l_i = l_i * alpha + rs;
  m_i = mnew;
  return P;
}

// ---------------------------------------------------------------------------
// 3) Flash attention.  grid = (16 q-groups, 32 b*h), block = 128 (4 waves).
//    Each wave: 2 query tiles (32 queries) x 2048 keys in blocks of 32.
//    K/V operand fetches are shared between both query tiles (2x reuse);
//    16 WMMAs per 32-key block.
// ---------------------------------------------------------------------------
__global__ __launch_bounds__(128) void flash_attn_kernel(
    const __bf16* __restrict__ qr, const __bf16* __restrict__ kr,
    const __bf16* __restrict__ vt, __bf16* __restrict__ ctx) {
  const int lane = threadIdx.x & 31;
  const int wave = threadIdx.x >> 5;
  const int bh = blockIdx.y;                      // b*16 + h
  const int lq = lane & 15, hi = lane >> 4, koff = hi * 8;
  const int q0 = (blockIdx.x * 4 + wave) * 32;

  // Q as B-operand (lane = query column), loop-invariant; two q-tiles
  const __bf16* qrowA = qr + ((size_t)bh * 2048 + q0 + lq) * 64;
  const __bf16* qrowB = qrowA + (size_t)16 * 64;
  v16bf BqA0 = mk16(qrowA + koff,      qrowA + 16 + koff);   // d 0..31
  v16bf BqA1 = mk16(qrowA + 32 + koff, qrowA + 48 + koff);   // d 32..63
  v16bf BqB0 = mk16(qrowB + koff,      qrowB + 16 + koff);
  v16bf BqB1 = mk16(qrowB + 32 + koff, qrowB + 48 + koff);

  const __bf16* kbase = kr + (size_t)bh * 2048 * 64;
  const __bf16* vbase = vt + (size_t)bh * 64 * 2048;         // [D,T]

  float mA = -3.0e38f, lA = 0.f, mB = -3.0e38f, lB = 0.f;
  v8f OA[4], OB[4];
#pragma unroll
  for (int dt = 0; dt < 4; ++dt) { OA[dt] = vzero8(); OB[dt] = vzero8(); }

  for (int kb = 0; kb < 2048; kb += 32) {
    // K operands, shared by both q-tiles
    const __bf16* k0p = kbase + (size_t)(kb + lq) * 64;
    const __bf16* k1p = k0p + (size_t)16 * 64;
    v16bf Ka0 = mk16(k0p + koff,      k0p + 16 + koff);
    v16bf Ka1 = mk16(k0p + 32 + koff, k0p + 48 + koff);
    v16bf Kb0 = mk16(k1p + koff,      k1p + 16 + koff);
    v16bf Kb1 = mk16(k1p + 32 + koff, k1p + 48 + koff);

    // S^T tiles (rows = keys, cols = queries)
    v8f sA0 = vzero8(), sA1 = vzero8(), sB0 = vzero8(), sB1 = vzero8();
    sA0 = WMMA_BF16(Ka0, BqA0, sA0);  sA0 = WMMA_BF16(Ka1, BqA1, sA0);
    sA1 = WMMA_BF16(Kb0, BqA0, sA1);  sA1 = WMMA_BF16(Kb1, BqA1, sA1);
    sB0 = WMMA_BF16(Ka0, BqB0, sB0);  sB0 = WMMA_BF16(Ka1, BqB1, sB0);
    sB1 = WMMA_BF16(Kb0, BqB0, sB1);  sB1 = WMMA_BF16(Kb1, BqB1, sB1);

    float aA, aB;
    v16bf Pa = softmax_update(sA0, sA1, mA, lA, aA);
    v16bf Pb = softmax_update(sB0, sB1, mB, lB, aB);

    // O^T accumulate; V operand loaded once per d-tile, used by both q-tiles
#pragma unroll
    for (int dt = 0; dt < 4; ++dt) {
      const __bf16* vp = vbase + (size_t)(dt * 16 + lq) * 2048 + kb;
      v16bf Vt = mk16(vp + koff, vp + 16 + koff);
#pragma unroll
      for (int v = 0; v < 8; ++v) OA[dt][v] *= aA;
      OA[dt] = WMMA_BF16(Vt, Pa, OA[dt]);
#pragma unroll
      for (int v = 0; v < 8; ++v) OB[dt][v] *= aB;
      OB[dt] = WMMA_BF16(Vt, Pb, OB[dt]);
    }
  }

  // normalize; store ctx[b, t, h*64 + d] as packed bf16 (b128 stores)
  const float invA = 1.f / lA, invB = 1.f / lB;
  const int b = bh >> 4, h = bh & 15;
  __bf16* crowA = ctx + (size_t)(b * 2048 + q0 + lq) * 1024 + h * 64 + hi * 8;
  __bf16* crowB = crowA + (size_t)16 * 1024;
#pragma unroll
  for (int dt = 0; dt < 4; ++dt) {
    v8bf o;
#pragma unroll
    for (int v = 0; v < 8; ++v) o[v] = (__bf16)(OA[dt][v] * invA);
    *reinterpret_cast<v8bf*>(crowA + dt * 16) = o;
#pragma unroll
    for (int v = 0; v < 8; ++v) o[v] = (__bf16)(OB[dt][v] * invB);
    *reinterpret_cast<v8bf*>(crowB + dt * 16) = o;
  }
}

// ---------------------------------------------------------------------------
// 4) Output projection: out = ctx @ wo^T + bo (f32), 2x2 WMMA blocking
// ---------------------------------------------------------------------------
__global__ __launch_bounds__(256) void out_proj_kernel(
    const __bf16* __restrict__ ctx, const __bf16* __restrict__ wob,
    const float* __restrict__ bo, float* __restrict__ out) {
  const int lane = threadIdx.x & 31;
  const int wave = threadIdx.x >> 5;
  const int mt = blockIdx.y * 8 + wave;
  const int nt = blockIdx.x;
  const int lq = lane & 15, hi = lane >> 4, koff = hi * 8;
  const int row0 = mt * 32, col0 = nt * 32;

  const __bf16* arow0 = ctx + (size_t)(row0 + lq) * 1024;
  const __bf16* arow1 = arow0 + (size_t)16 * 1024;
  const __bf16* brow0 = wob + (size_t)(col0 + lq) * 1024;
  const __bf16* brow1 = brow0 + (size_t)16 * 1024;

  v8f acc[2][2];
  acc[0][0] = vzero8(); acc[0][1] = vzero8();
  acc[1][0] = vzero8(); acc[1][1] = vzero8();

#pragma unroll 2
  for (int k0 = 0; k0 < 1024; k0 += 32) {
    __builtin_prefetch(arow0 + k0 + 256, 0, 1);
    __builtin_prefetch(brow0 + k0 + 256, 0, 1);
    v16bf A0 = mk16(arow0 + k0 + koff, arow0 + k0 + 16 + koff);
    v16bf A1 = mk16(arow1 + k0 + koff, arow1 + k0 + 16 + koff);
    v16bf B0 = mk16(brow0 + k0 + koff, brow0 + k0 + 16 + koff);
    v16bf B1 = mk16(brow1 + k0 + koff, brow1 + k0 + 16 + koff);
    acc[0][0] = WMMA_BF16(A0, B0, acc[0][0]);
    acc[0][1] = WMMA_BF16(A0, B1, acc[0][1]);
    acc[1][0] = WMMA_BF16(A1, B0, acc[1][0]);
    acc[1][1] = WMMA_BF16(A1, B1, acc[1][1]);
  }

#pragma unroll
  for (int ni = 0; ni < 2; ++ni) {
    const int   c    = col0 + ni * 16 + lq;
    const float bval = bo[c];
#pragma unroll
    for (int mi = 0; mi < 2; ++mi) {
      const int rbase = row0 + mi * 16;
#pragma unroll
      for (int v = 0; v < 8; ++v)
        out[(size_t)(rbase + v + hi * 8) * 1024 + c] = acc[mi][ni][v] + bval;
    }
  }
}

// ---------------------------------------------------------------------------
// Host launcher
// ---------------------------------------------------------------------------
extern "C" void kernel_launch(void* const* d_in, const int* in_sizes, int n_in,
                              void* d_out, int out_size, void* d_ws, size_t ws_size,
                              hipStream_t stream) {
  (void)in_sizes; (void)n_in; (void)out_size; (void)ws_size;
  const float* x  = (const float*)d_in[0];
  const float* wq = (const float*)d_in[1];
  const float* bq = (const float*)d_in[2];
  const float* wk = (const float*)d_in[3];
  const float* bk = (const float*)d_in[4];
  const float* wv = (const float*)d_in[5];
  const float* bv = (const float*)d_in[6];
  const float* wo = (const float*)d_in[7];
  const float* bo = (const float*)d_in[8];
  float* out = (float*)d_out;

  // workspace layout (bf16 elements): 48 MB total
  const size_t N_X = (size_t)4096 * 1024;         // x / ctx
  const size_t N_W = (size_t)1024 * 1024;         // each weight
  const size_t N_Q = (size_t)2 * 16 * 2048 * 64;  // q / k / vt
  __bf16* base = (__bf16*)d_ws;
  __bf16* xb  = base;
  __bf16* wqb = xb  + N_X;
  __bf16* wkb = wqb + N_W;
  __bf16* wvb = wkb + N_W;
  __bf16* wob = wvb + N_W;
  __bf16* qr  = wob + N_W;
  __bf16* kr  = qr  + N_Q;
  __bf16* vt  = kr  + N_Q;
  __bf16* ctx = vt  + N_Q;

  // 1) casts
  {
    int nv = (int)(N_X / 8);
    f32_to_bf16_vec8<<<(nv + 255) / 256, 256, 0, stream>>>(x, xb, nv);
    int nw = (int)(N_W / 8);
    f32_to_bf16_vec8<<<(nw + 255) / 256, 256, 0, stream>>>(wq, wqb, nw);
    f32_to_bf16_vec8<<<(nw + 255) / 256, 256, 0, stream>>>(wk, wkb, nw);
    f32_to_bf16_vec8<<<(nw + 255) / 256, 256, 0, stream>>>(wv, wvb, nw);
    f32_to_bf16_vec8<<<(nw + 255) / 256, 256, 0, stream>>>(wo, wob, nw);
  }

  // 2) QKV + RoPE: 32 n-blocks x 16 m-groups x {q,k,v}, 32x32 per wave
  qkv_rope_kernel<<<dim3(32, 16, 3), 256, 0, stream>>>(
      xb, wqb, wkb, wvb, bq, bk, bv, qr, kr, vt);

  // 3) flash attention: 16 q-groups (4 waves x 32 queries) x 32 (b,h)
  flash_attn_kernel<<<dim3(16, 32), 128, 0, stream>>>(qr, kr, vt, ctx);

  // 4) output projection: 32x32 per wave
  out_proj_kernel<<<dim3(32, 16), 256, 0, stream>>>(ctx, wob, bo, out);
}
